// LinearCrossAttention_60988535603243
// MI455X (gfx1250) — compile-verified, matrix operand order checked by reading, not run on hardware
//
#include <hip/hip_runtime.h>
#include <hip/hip_bf16.h>

typedef __attribute__((ext_vector_type(16))) _Float16 v16h;
typedef __attribute__((ext_vector_type(8)))  _Float16 v8h;
typedef __attribute__((ext_vector_type(8)))  float    v8f;

#define DMODEL 1024
#define NHEADS 16
#define NKV    4
#define DHEAD  64
#define NREP   4
#define BATCH  4
#define TGTN   2048
#define SRCN   4096
#define MAXD   16
#define KVDIM  (NKV * DHEAD)   // 256
#define EPSV   1e-5f

#define SHUF16(lo, hi) __builtin_shufflevector((lo), (hi), 0,1,2,3,4,5,6,7,8,9,10,11,12,13,14,15)

__device__ __forceinline__ float gelu1p(float x) {
    return 0.5f * x * (1.0f + erff(x * 0.70710678118654752f)) + 1.0f;
}

__device__ __forceinline__ v8f wmma_f16(v16h a, v16h b, v8f c) {
    return __builtin_amdgcn_wmma_f32_16x16x32_f16(false, a, false, b, (short)0, c, false, false);
}

// gfx1250 async global->LDS copy (ASYNCcnt path), 16 bytes per lane
__device__ __forceinline__ void async_copy_b128(const void* gsrc, void* lds_dst) {
    unsigned int l = (unsigned int)(unsigned long long)lds_dst;   // LDS aperture: low 32 bits = LDS offset
    asm volatile("global_load_async_to_lds_b128 %0, %1, off"
                 :: "v"(l), "v"(gsrc) : "memory");
}
__device__ __forceinline__ void wait_async0() {
    asm volatile("s_wait_asynccnt 0x0" ::: "memory");
}

__device__ __forceinline__ void cvt16_f32_to_f16(const float* Ap, v8h& h0, v8h& h1) {
    float4 f0 = *(const float4*)(Ap + 0);
    float4 f1 = *(const float4*)(Ap + 4);
    float4 f2 = *(const float4*)(Ap + 8);
    float4 f3 = *(const float4*)(Ap + 12);
    h0[0]=(_Float16)f0.x; h0[1]=(_Float16)f0.y; h0[2]=(_Float16)f0.z; h0[3]=(_Float16)f0.w;
    h0[4]=(_Float16)f1.x; h0[5]=(_Float16)f1.y; h0[6]=(_Float16)f1.z; h0[7]=(_Float16)f1.w;
    h1[0]=(_Float16)f2.x; h1[1]=(_Float16)f2.y; h1[2]=(_Float16)f2.z; h1[3]=(_Float16)f2.w;
    h1[4]=(_Float16)f3.x; h1[5]=(_Float16)f3.y; h1[6]=(_Float16)f3.z; h1[7]=(_Float16)f3.w;
}

// ---------------------------------------------------------------------------
// f32 -> f16 conversion (weights)
// ---------------------------------------------------------------------------
__global__ void cvt_f32_to_f16(const float* __restrict__ in, _Float16* __restrict__ out, int n) {
    int i = blockIdx.x * 256 + threadIdx.x;
    if (i < n) out[i] = (_Float16)in[i];
}

// ---------------------------------------------------------------------------
// per-batch doc segment offsets: out[b][d] = lower_bound(ids[b,:], d)
// ---------------------------------------------------------------------------
__global__ void lower_bounds_kernel(const int* __restrict__ ids, int len, int* __restrict__ out) {
    const int b = blockIdx.x;
    const int d = threadIdx.x;
    if (d > MAXD) return;
    const int* p = ids + (size_t)b * len;
    int lo = 0, hi = len;
    while (lo < hi) {
        int mid = (lo + hi) >> 1;
        if (p[mid] < d) lo = mid + 1; else hi = mid;
    }
    out[b * (MAXD + 1) + d] = lo;
}

// ---------------------------------------------------------------------------
// Tiled WMMA GEMM: C[M,N] = A[M,K] * B[K,N]
//   Tile 128x64, K-step 64, 128 threads (4 waves); each wave: 32-row strip,
//   16 WMMAs per staged tile (2 row-tiles x 4 col-tiles x 2 k-halves).
//   A: f32 (converted while staging) or f16 (async global->LDS)
//   B: f16 row-major, staged TRANSPOSED (BsT[n][k], stride 72 halfs)
//   EPI 0: f32 raw; 1: gelu1p->f16; 2: gelu1p*mask->f16; 3: mask->f16
// ---------------------------------------------------------------------------
template<int EPI, bool AF32>
__global__ __launch_bounds__(128)
void gemm_wmma(const void* __restrict__ Av, const _Float16* __restrict__ Bw,
               void* __restrict__ Cv, int M, int N, int K,
               const unsigned char* __restrict__ mask)
{
    __shared__ _Float16 As[128 * 64];      // row-major: A frags contiguous
    __shared__ _Float16 BsT[64 * 72];      // BsT[n][k], padded stride 72 (16B aligned)

    const int tid  = threadIdx.x;
    const int wave = tid >> 5;
    const int lane = tid & 31;
    const int lm   = lane & 15;
    const int kb   = (lane < 16) ? 0 : 8;
    const int bm   = blockIdx.x * 128;
    const int bn   = blockIdx.y * 64;

    v8f acc[2][4] = {};

    for (int k0 = 0; k0 < K; k0 += 64) {
        // ---- stage A tile: one full 64-elem row per thread ----
        if (AF32) {
            const float* Ap = (const float*)Av + (size_t)(bm + tid) * K + k0;
            #pragma unroll
            for (int c = 0; c < 4; ++c) {
                v8h h0, h1;
                cvt16_f32_to_f16(Ap + c * 16, h0, h1);
                *(v8h*)&As[tid * 64 + c * 16]     = h0;
                *(v8h*)&As[tid * 64 + c * 16 + 8] = h1;
            }
        } else {
            const _Float16* Ap = (const _Float16*)Av + (size_t)(bm + tid) * K + k0;
            #pragma unroll
            for (int j = 0; j < 8; ++j) async_copy_b128(Ap + j * 8, &As[tid * 64 + j * 8]);
        }
        // ---- stage B tile, transposed: thread -> (k-row, 32-col half) ----
        {
            const int krow = tid >> 1, c0 = (tid & 1) * 32;
            const _Float16* Bp = Bw + (size_t)(k0 + krow) * N + bn + c0;
            v8h b0 = *(const v8h*)Bp;
            v8h b1 = *(const v8h*)(Bp + 8);
            v8h b2 = *(const v8h*)(Bp + 16);
            v8h b3 = *(const v8h*)(Bp + 24);
            #pragma unroll
            for (int i = 0; i < 8; ++i) {
                BsT[(c0 + i) * 72 + krow]      = b0[i];
                BsT[(c0 + 8 + i) * 72 + krow]  = b1[i];
                BsT[(c0 + 16 + i) * 72 + krow] = b2[i];
                BsT[(c0 + 24 + i) * 72 + krow] = b3[i];
            }
        }
        if (!AF32) wait_async0();
        __syncthreads();

        // A fragments: 2 row-tiles x 2 k-halves, contiguous b128 pairs
        v16h a[2][2];
        #pragma unroll
        for (int rt = 0; rt < 2; ++rt) {
            const _Float16* ar = &As[(32 * wave + 16 * rt + lm) * 64];
            a[rt][0] = SHUF16(*(const v8h*)(ar + kb),      *(const v8h*)(ar + 16 + kb));
            a[rt][1] = SHUF16(*(const v8h*)(ar + 32 + kb), *(const v8h*)(ar + 48 + kb));
        }
        #pragma unroll
        for (int t = 0; t < 4; ++t) {
            const _Float16* br = &BsT[(t * 16 + lm) * 72];
            v16h bf0 = SHUF16(*(const v8h*)(br + kb),      *(const v8h*)(br + 16 + kb));
            v16h bf1 = SHUF16(*(const v8h*)(br + 32 + kb), *(const v8h*)(br + 48 + kb));
            #pragma unroll
            for (int rt = 0; rt < 2; ++rt) {
                acc[rt][t] = wmma_f16(a[rt][0], bf0, acc[rt][t]);
                acc[rt][t] = wmma_f16(a[rt][1], bf1, acc[rt][t]);
            }
        }
        __syncthreads();
    }

    // ---- epilogue ----
    const int moff = (lane < 16) ? 0 : 8;
    #pragma unroll
    for (int rt = 0; rt < 2; ++rt) {
        #pragma unroll
        for (int t = 0; t < 4; ++t) {
            #pragma unroll
            for (int r = 0; r < 8; ++r) {
                const int m = bm + 32 * wave + 16 * rt + moff + r;
                const int n = bn + t * 16 + lm;
                float v = acc[rt][t][r];
                if (EPI == 0) {
                    ((float*)Cv)[(size_t)m * N + n] = v;
                } else {
                    if (EPI == 1 || EPI == 2) v = gelu1p(v);
                    if (EPI == 2 || EPI == 3) v *= (mask[m] != 0) ? 1.0f : 0.0f;
                    ((_Float16*)Cv)[(size_t)m * N + n] = (_Float16)v;
                }
            }
        }
    }
}

// ---------------------------------------------------------------------------
// Per (b, kv-head g, doc d): kv[d][e] = sum_s K[s][d] V[s][e],  z[d] = sum_s K[s][d]
// Both K^T chunk and V chunk staged transposed => all frag loads contiguous.
// Output stored TRANSPOSED: kvT[e][d]  (so numden B-frag loads are contiguous).
// ---------------------------------------------------------------------------
__global__ __launch_bounds__(128)
void kv_kernel(const _Float16* __restrict__ k16, const _Float16* __restrict__ v16,
               const int* __restrict__ srcOff,
               _Float16* __restrict__ kvT, float* __restrict__ zOut)
{
    __shared__ _Float16 KsT[64 * 40];   // KsT[m][s] : m = d_head idx of K
    __shared__ _Float16 VsT[64 * 40];   // VsT[e][s]

    const int d = blockIdx.x & 15;
    const int g = (blockIdx.x >> 4) & 3;
    const int b = blockIdx.x >> 6;
    const int tid  = threadIdx.x;
    const int wave = tid >> 5;
    const int lane = tid & 31;
    const int lm   = lane & 15;
    const int kb   = (lane < 16) ? 0 : 8;

    const int s0 = srcOff[b * (MAXD + 1) + d];
    const int s1 = srcOff[b * (MAXD + 1) + d + 1];

    v8f acc[4] = {};
    float zreg = 0.0f;

    const int row = tid >> 2, col = (tid & 3) * 16;   // row = s within chunk

    for (int sc = s0; sc < s1; sc += 32) {
        const int s = sc + row;
        v8h k0v = {}, k1v = {}, v0v = {}, v1v = {};
        if (s < s1) {
            const _Float16* kp = k16 + (size_t)(b * SRCN + s) * KVDIM + g * DHEAD + col;
            const _Float16* vp = v16 + (size_t)(b * SRCN + s) * KVDIM + g * DHEAD + col;
            k0v = *(const v8h*)kp; k1v = *(const v8h*)(kp + 8);
            v0v = *(const v8h*)vp; v1v = *(const v8h*)(vp + 8);
        }
        #pragma unroll
        for (int i = 0; i < 8; ++i) {
            KsT[(col + i) * 40 + row]     = k0v[i];
            KsT[(col + 8 + i) * 40 + row] = k1v[i];
            VsT[(col + i) * 40 + row]     = v0v[i];
            VsT[(col + 8 + i) * 40 + row] = v1v[i];
        }
        __syncthreads();

        if (tid < 64) {  // z: contiguous row of KsT
            #pragma unroll
            for (int c = 0; c < 4; ++c) {
                v8h z8 = *(const v8h*)&KsT[tid * 40 + c * 8];
                #pragma unroll
                for (int i = 0; i < 8; ++i) zreg += (float)z8[i];
            }
        }

        v16h a;
        {
            const _Float16* ar = &KsT[(16 * wave + lm) * 40];
            a = SHUF16(*(const v8h*)(ar + kb), *(const v8h*)(ar + 16 + kb));
        }
        #pragma unroll
        for (int t = 0; t < 4; ++t) {
            const _Float16* br = &VsT[(t * 16 + lm) * 40];
            v16h bf = SHUF16(*(const v8h*)(br + kb), *(const v8h*)(br + 16 + kb));
            acc[t] = wmma_f16(a, bf, acc[t]);
        }
        __syncthreads();
    }

    const size_t base = (size_t)((b * NKV + g) * MAXD + d) * (DHEAD * DHEAD);
    const int moff = (lane < 16) ? 0 : 8;
    #pragma unroll
    for (int t = 0; t < 4; ++t)
        #pragma unroll
        for (int r = 0; r < 8; ++r) {
            const int m = 16 * wave + moff + r;   // d index
            const int n = t * 16 + lm;            // e index
            kvT[base + (size_t)n * 64 + m] = (_Float16)acc[t][r];   // transposed store
        }
    if (tid < 64) zOut[(size_t)((b * NKV + g) * MAXD + d) * 64 + tid] = zreg;
}

// ---------------------------------------------------------------------------
// num/den: per (b, 16-token tile, kv-group g), 4 waves = 4 heads of group g.
//   C += row_mask(q_tile) @ kv_d (WMMA),  den[t] += q[t]·z_d (VALU)
// kvT/q tiles staged via async global->LDS. attn = num/(den+eps) as f16.
// ---------------------------------------------------------------------------
__global__ __launch_bounds__(128)
void numden_kernel(const _Float16* __restrict__ q16,
                   const _Float16* __restrict__ kvT,
                   const float* __restrict__ z32,
                   const int* __restrict__ tgtDoc,
                   _Float16* __restrict__ attn)
{
    __shared__ _Float16 qs[4][16 * 64];   // per-wave q tile, row-major
    __shared__ _Float16 kvsT[64 * 72];    // kvsT[e][d], padded stride 72 (16B aligned)
    __shared__ float    zs[64];
    __shared__ int      docs[16];
    __shared__ float    dens[4][16];

    const int TILES = TGTN / 16;
    const int tile = blockIdx.x % TILES;
    const int g    = (blockIdx.x / TILES) % NKV;
    const int b    = blockIdx.x / (TILES * NKV);
    const int t0   = tile * 16;

    const int tid  = threadIdx.x;
    const int wave = tid >> 5;
    const int lane = tid & 31;
    const int lm   = lane & 15;
    const int kb   = (lane < 16) ? 0 : 8;
    const int h    = g * NREP + wave;

    if (tid < 16) docs[tid] = tgtDoc[b * TGTN + t0 + tid];

    {   // async stage this wave's q tile (16 x 64 f16)
        const int row = lane >> 1, c0 = (lane & 1) * 32;
        const _Float16* qp = q16 + (size_t)(b * TGTN + t0 + row) * DMODEL + h * DHEAD + c0;
        _Float16* dst = &qs[wave][row * 64 + c0];
        #pragma unroll
        for (int j = 0; j < 4; ++j) async_copy_b128(qp + j * 8, dst + j * 8);
    }
    wait_async0();
    __syncthreads();
    const int dmin = docs[0], dmax = docs[15];   // sorted ids -> contiguous doc range
    const int myDoc = docs[lm];

    v16h a0, a1;
    {
        const _Float16* ar = &qs[wave][lm * 64];
        a0 = SHUF16(*(const v8h*)(ar + kb),      *(const v8h*)(ar + 16 + kb));
        a1 = SHUF16(*(const v8h*)(ar + 32 + kb), *(const v8h*)(ar + 48 + kb));
    }

    v8f acc[4] = {};
    float den = 0.0f;

    for (int d = dmin; d <= dmax; ++d) {
        __syncthreads();
        {   // async stage kv_d^T (shared by all 4 waves)
            const int row = tid >> 1, c0 = (tid & 1) * 32;
            const _Float16* kp = kvT + (size_t)((b * NKV + g) * MAXD + d) * (DHEAD * DHEAD)
                               + (size_t)row * 64 + c0;
            _Float16* dst = &kvsT[row * 72 + c0];
            #pragma unroll
            for (int j = 0; j < 4; ++j) async_copy_b128(kp + j * 8, dst + j * 8);
        }
        if (tid < 64) zs[tid] = z32[(size_t)((b * NKV + g) * MAXD + d) * 64 + tid];
        wait_async0();
        __syncthreads();

        // per-lane row mask (EXEC stays full for WMMA)
        const bool on = (myDoc == d);
        v16h m0, m1;
        #pragma unroll
        for (int i = 0; i < 16; ++i) {
            m0[i] = on ? a0[i] : (_Float16)0.0f;
            m1[i] = on ? a1[i] : (_Float16)0.0f;
        }

        #pragma unroll
        for (int t = 0; t < 4; ++t) {
            const _Float16* br = &kvsT[(t * 16 + lm) * 72];
            v16h bf0 = SHUF16(*(const v8h*)(br + kb),      *(const v8h*)(br + 16 + kb));
            acc[t] = wmma_f16(m0, bf0, acc[t]);
            v16h bf1 = SHUF16(*(const v8h*)(br + 32 + kb), *(const v8h*)(br + 48 + kb));
            acc[t] = wmma_f16(m1, bf1, acc[t]);
        }

        if (lane < 16 && on) {
            const _Float16* qr = &qs[wave][lane * 64];
            float s = 0.0f;
            #pragma unroll
            for (int e = 0; e < 64; ++e) s += (float)qr[e] * zs[e];
            den += s;
        }
    }

    if (lane < 16) dens[wave][lane] = den;
    __syncthreads();

    const int moff = (lane < 16) ? 0 : 8;
    #pragma unroll
    for (int t = 0; t < 4; ++t)
        #pragma unroll
        for (int r = 0; r < 8; ++r) {
            const int trow  = moff + r;
            const float dv  = dens[wave][trow] + EPSV;
            const float v   = acc[t][r] / dv;
            const int token = t0 + trow;
            attn[(size_t)(b * TGTN + token) * DMODEL + h * DHEAD + t * 16 + lm] = (_Float16)v;
        }
}

// ---------------------------------------------------------------------------
// launch
// ---------------------------------------------------------------------------
extern "C" void kernel_launch(void* const* d_in, const int* in_sizes, int n_in,
                              void* d_out, int out_size, void* d_ws, size_t ws_size,
                              hipStream_t stream)
{
    (void)in_sizes; (void)n_in; (void)out_size;

    const float* x    = (const float*)d_in[0];
    const float* enc  = (const float*)d_in[1];
    const float* Wq   = (const float*)d_in[2];
    const float* Wk   = (const float*)d_in[3];
    const float* Wv   = (const float*)d_in[4];
    const float* Wo   = (const float*)d_in[5];
    const unsigned char* mask = (const unsigned char*)d_in[6];
    const int*   srcDoc = (const int*)d_in[7];
    const int*   tgtDoc = (const int*)d_in[8];

    char* ws = (char*)d_ws;
    size_t off = 0;
    auto take = [&](size_t bytes) -> void* {
        void* p = (void*)(ws + off);
        off += (bytes + 255) & ~(size_t)255;
        return p;
    };

    _Float16* Wq16   = (_Float16*)take((size_t)DMODEL * DMODEL * 2);
    _Float16* Wk16   = (_Float16*)take((size_t)DMODEL * KVDIM * 2);
    _Float16* Wv16   = (_Float16*)take((size_t)DMODEL * KVDIM * 2);
    _Float16* Wo16   = (_Float16*)take((size_t)DMODEL * DMODEL * 2);
    _Float16* q16    = (_Float16*)take((size_t)BATCH * TGTN * DMODEL * 2);
    _Float16* k16    = (_Float16*)take((size_t)BATCH * SRCN * KVDIM * 2);
    _Float16* v16    = (_Float16*)take((size_t)BATCH * SRCN * KVDIM * 2);
    _Float16* attn16 = (_Float16*)take((size_t)BATCH * TGTN * DMODEL * 2);
    _Float16* kvT    = (_Float16*)take((size_t)BATCH * NKV * MAXD * DHEAD * DHEAD * 2);
    float*    z32    = (float*)take((size_t)BATCH * NKV * MAXD * DHEAD * 4);
    int*      srcOff = (int*)take((size_t)BATCH * (MAXD + 1) * 4);
    if (off > ws_size) return;

    // 1) weights -> f16
    {
        int n = DMODEL * DMODEL;
        cvt_f32_to_f16<<<(n + 255) / 256, 256, 0, stream>>>(Wq, Wq16, n);
        cvt_f32_to_f16<<<(n + 255) / 256, 256, 0, stream>>>(Wo, Wo16, n);
        int m = DMODEL * KVDIM;
        cvt_f32_to_f16<<<(m + 255) / 256, 256, 0, stream>>>(Wk, Wk16, m);
        cvt_f32_to_f16<<<(m + 255) / 256, 256, 0, stream>>>(Wv, Wv16, m);
    }

    // 2) doc segment offsets (src side)
    lower_bounds_kernel<<<BATCH, 32, 0, stream>>>(srcDoc, SRCN, srcOff);

    // 3) projections
    {
        dim3 gq(BATCH * TGTN / 128, DMODEL / 64);
        gemm_wmma<1, true><<<gq, 128, 0, stream>>>(x, Wq16, q16,
            BATCH * TGTN, DMODEL, DMODEL, nullptr);                 // Q = gelu1p(x Wq)
        dim3 gk(BATCH * SRCN / 128, KVDIM / 64);
        gemm_wmma<2, true><<<gk, 128, 0, stream>>>(enc, Wk16, k16,
            BATCH * SRCN, KVDIM, DMODEL, mask);                     // K = gelu1p(enc Wk)*mask
        gemm_wmma<3, true><<<gk, 128, 0, stream>>>(enc, Wv16, v16,
            BATCH * SRCN, KVDIM, DMODEL, mask);                     // V = (enc Wv)*mask
    }

    // 4) per-doc kv / z
    kv_kernel<<<BATCH * NKV * MAXD, 128, 0, stream>>>(k16, v16, srcOff, kvT, z32);

    // 5) num/den -> attn (f16)
    numden_kernel<<<BATCH * NKV * (TGTN / 16), 128, 0, stream>>>(q16, kvT, z32, tgtDoc, attn16);

    // 6) output projection -> d_out (f32)
    {
        dim3 go(BATCH * TGTN / 128, DMODEL / 64);
        gemm_wmma<0, false><<<go, 128, 0, stream>>>(attn16, Wo16, (float*)d_out,
            BATCH * TGTN, DMODEL, DMODEL, nullptr);
    }
}
